// DGLJTNNDecoder_2379411882640
// MI455X (gfx1250) — matrix operand order repaired and checked by reading, not compiled, not run on hardware
//
#include <hip/hip_runtime.h>
#include <hip/hip_bf16.h>
#include <math.h>
#include <stdint.h>

// ---------------------------------------------------------------------------
// JTNN decoder forward pass for MI455X (gfx1250, wave32).
// GEMMs: v_wmma_f32_16x16x32_bf16 (f32 accumulate).
// GRU weight panels are staged into LDS via the Tensor Data Mover
// (tensor_load_to_lds, TENSORcnt) and consumed with ds_load_b128.
// ---------------------------------------------------------------------------

typedef __bf16 bf16_t;
typedef __attribute__((ext_vector_type(16))) __bf16 v16bf;
typedef __attribute__((ext_vector_type(8)))  __bf16 v8bf;
typedef __attribute__((ext_vector_type(8)))  float  v8f;
typedef __attribute__((ext_vector_type(4)))  unsigned int u32x4;
typedef __attribute__((ext_vector_type(8)))  int  i32x8;
typedef __attribute__((ext_vector_type(4)))  int  i32x4;

#define B_N   4096
#define L_N   16
#define H_N   256
#define LAT_N 64
#define V_N   800
#define STEP  (B_N * H_N)          // elements per [B,H] slab

__device__ __forceinline__ v8f vzero8f() {
  v8f z = {0.f, 0.f, 0.f, 0.f, 0.f, 0.f, 0.f, 0.f};
  return z;
}

__device__ __forceinline__ v8f wmma_bf16(v16bf a, v16bf b, v8f c) {
  return __builtin_amdgcn_wmma_f32_16x16x32_bf16(
      /*neg_a=*/false, a, /*neg_b=*/false, b,
      /*c_mod=*/(short)0, c, /*reuse_a=*/false, /*reuse_b=*/false);
}

// A fragment: 16x32 bf16 tile from a row-major [rows, lda] bf16 matrix.
__device__ __forceinline__ v16bf load_a_frag(const bf16_t* __restrict__ tile,
                                             int lda, int lane) {
  int row  = lane & 15;
  int half = lane >> 4;
  const bf16_t* p = tile + (size_t)row * lda + half * 8;
  v8bf lo = *(const v8bf*)p;
  v8bf hi = *(const v8bf*)(p + 16);
  v16bf a;
#pragma unroll
  for (int i = 0; i < 8; ++i) { a[i] = lo[i]; a[i + 8] = hi[i]; }
  return a;
}

// B fragment from packed weights in global memory.
__device__ __forceinline__ v16bf load_b_frag(const bf16_t* __restrict__ packed,
                                             int kt, int nt, int NT, int lane) {
  const bf16_t* p = packed + (((size_t)(kt * NT + nt)) * 32 + lane) * 16;
  v8bf lo = *(const v8bf*)p;
  v8bf hi = *(const v8bf*)(p + 8);
  v16bf b;
#pragma unroll
  for (int i = 0; i < 8; ++i) { b[i] = lo[i]; b[i + 8] = hi[i]; }
  return b;
}

// B fragment from an LDS panel of [8 kt][4 nt][32 lanes][16] bf16.
__device__ __forceinline__ v16bf load_b_lds(const bf16_t* base, int kt, int j,
                                            int lane) {
  const bf16_t* p = base + (((kt * 4 + j) * 32 + lane) * 16);
  v8bf lo = *(const v8bf*)p;
  v8bf hi = *(const v8bf*)(p + 8);
  v16bf b;
#pragma unroll
  for (int i = 0; i < 8; ++i) { b[i] = lo[i]; b[i + 8] = hi[i]; }
  return b;
}

// ---------------------------------------------------------------------------
// Tensor Data Mover: DMA a 2D tile (nrows rows of row_dw dwords, global row
// stride stride_dw dwords) from global memory into LDS.  D# per CDNA5 ISA:
// group0 = {count=1, lds_addr, global_addr, type=2};
// group1 = {data_size=4B, tensor/tile dims, dim0 stride}.
// ---------------------------------------------------------------------------
__device__ __forceinline__ void tdm_load_2d(unsigned lds_byte_addr,
                                            const void* gptr,
                                            unsigned row_dw, unsigned nrows,
                                            unsigned stride_dw) {
  unsigned long long ga = (unsigned long long)(uintptr_t)gptr;
  u32x4 g0;
  g0[0] = 1u;                                                // count=1 (valid)
  g0[1] = lds_byte_addr;                                     // lds_addr
  g0[2] = (unsigned)ga;                                      // global_addr lo
  g0[3] = (unsigned)((ga >> 32) & 0x01FFFFFFu) | (2u << 30); // addr hi | type=2
  i32x8 g1;
  g1[0] = 0x00020000;                                    // data_size = 4B
  g1[1] = (int)((row_dw & 0xFFFFu) << 16);               // tensor_dim0 lo16
  g1[2] = (int)(((row_dw >> 16) & 0xFFFFu) |             // tensor_dim0 hi16
                ((nrows & 0xFFFFu) << 16));              // tensor_dim1 lo16
  g1[3] = (int)(((nrows >> 16) & 0xFFFFu) |              // tensor_dim1 hi16
                ((row_dw & 0xFFFFu) << 16));             // tile_dim0
  g1[4] = (int)(nrows & 0xFFFFu);                        // tile_dim1 (dim2=0)
  g1[5] = (int)stride_dw;                                // dim0_stride lo32
  g1[6] = 0;
  g1[7] = 0;
  i32x4 z4 = {0, 0, 0, 0};
#if defined(__clang_major__) && __clang_major__ >= 23
  i32x8 z8 = {0, 0, 0, 0, 0, 0, 0, 0};
  __builtin_amdgcn_tensor_load_to_lds(g0, g1, z4, z4, z8, 0);
#else
  __builtin_amdgcn_tensor_load_to_lds(g0, g1, z4, z4, 0);
#endif
}

// ---------------------------------------------------------------------------
// Prep kernels
// ---------------------------------------------------------------------------

__global__ void pack_w_kernel(const float* __restrict__ W,
                              bf16_t* __restrict__ dst, int N, int K) {
  int idx = blockIdx.x * 256 + threadIdx.x;
  if (idx >= N * K) return;
  int i    = idx & 15;
  int lane = (idx >> 4) & 31;
  int tile = idx >> 9;
  int NT   = N >> 4;
  int nt   = tile % NT;
  int kt   = tile / NT;
  int n    = nt * 16 + (lane & 15);
  int k    = kt * 32 + ((lane >> 4) << 3) + (i & 7) + ((i >> 3) << 4);
  dst[idx] = (bf16_t)W[(size_t)n * K + k];
}

__global__ void embed_kernel(const int* __restrict__ wid,
                             const float* __restrict__ emb,
                             bf16_t* __restrict__ x) {
  int idx = blockIdx.x * 256 + threadIdx.x;   // L*B*H total, exact
  int h = idx & (H_N - 1);
  int b = (idx >> 8) & (B_N - 1);
  int t = idx >> 20;
  x[idx] = (bf16_t)emb[(size_t)wid[b * L_N + t] * H_N + h];
}

__global__ void cvt_kernel(const float* __restrict__ src,
                           bf16_t* __restrict__ dst, int n) {
  int idx = blockIdx.x * 256 + threadIdx.x;
  if (idx < n) dst[idx] = (bf16_t)src[idx];
}

__global__ void hrev_kernel(const bf16_t* __restrict__ mfwd,
                            const bf16_t* __restrict__ mrev,
                            bf16_t* __restrict__ hrev, int n) {
  int idx = blockIdx.x * 256 + threadIdx.x;
  if (idx >= n) return;
  float a = (idx >= STEP) ? (float)mfwd[idx - STEP] : 0.f;
  hrev[idx] = (bf16_t)(a + (float)mrev[idx]);
}

__global__ void init_out_kernel(float* out) {
  if (threadIdx.x < 4) out[threadIdx.x] = 0.f;
}

__global__ void final_kernel(float* out) {
  if (threadIdx.x == 0) {
    out[0] *= (1.f / (float)B_N);
    out[1] *= (1.f / (float)B_N);
    out[2] *= (1.f / (float)(L_N * B_N));
    out[3] *= (1.f / (float)((2 * L_N - 1) * B_N));
  }
}

// ---------------------------------------------------------------------------
// GRU step kernels with TDM-staged weight panels.
// ---------------------------------------------------------------------------

__global__ __launch_bounds__(256) void gru_zh_kernel(
    const bf16_t* __restrict__ xsrc, const bf16_t* __restrict__ mprev,
    const bf16_t* __restrict__ rmprev,
    const bf16_t* __restrict__ WzP, const bf16_t* __restrict__ WhP,
    const float* __restrict__ bz, const float* __restrict__ bh,
    bf16_t* __restrict__ mout) {
  extern __shared__ char smem[];
  bf16_t* lz = (bf16_t*)smem;                 // 32KB: [8 kt][4 nt][32][16]
  bf16_t* lh = (bf16_t*)(smem + 32768);       // 32KB
  int lane = threadIdx.x;
  int wv   = threadIdx.y;
  int cg   = blockIdx.x & 3;
  int row0 = ((blockIdx.x >> 2) * 8 + wv) * 16;
  unsigned lzA = (unsigned)(uintptr_t)(void*)lz;
  unsigned lhA = (unsigned)(uintptr_t)(void*)lh;

  v8f accZ[4], accH[4];
#pragma unroll
  for (int j = 0; j < 4; ++j) { accZ[j] = vzero8f(); accH[j] = vzero8f(); }

  // Phase 0: x_src against weight K-rows 0..7 of this column group.
  if (wv == 0) {
    tdm_load_2d(lzA, WzP + (size_t)(cg * 4) * 512, 1024, 8, 4096);
    tdm_load_2d(lhA, WhP + (size_t)(cg * 4) * 512, 1024, 8, 4096);
  }
  __builtin_amdgcn_s_wait_tensorcnt(0);
  __syncthreads();

  const bf16_t* xa = xsrc + (size_t)row0 * H_N;
  for (int kt = 0; kt < 8; ++kt) {
    v16bf a = load_a_frag(xa + kt * 32, H_N, lane);
#pragma unroll
    for (int j = 0; j < 4; ++j) {
      accZ[j] = wmma_bf16(a, load_b_lds(lz, kt, j, lane), accZ[j]);
      accH[j] = wmma_bf16(a, load_b_lds(lh, kt, j, lane), accH[j]);
    }
  }

  // Phase 1: carry part against weight K-rows 8..15; skipped at t=0.
  if (mprev) {
    __syncthreads();                          // phase-0 reads complete
    if (wv == 0) {
      tdm_load_2d(lzA, WzP + (size_t)(8 * 16 + cg * 4) * 512, 1024, 8, 4096);
      tdm_load_2d(lhA, WhP + (size_t)(8 * 16 + cg * 4) * 512, 1024, 8, 4096);
    }
    __builtin_amdgcn_s_wait_tensorcnt(0);
    __syncthreads();

    const bf16_t* ma = mprev  + (size_t)row0 * H_N;
    const bf16_t* ra = rmprev + (size_t)row0 * H_N;
    for (int kt = 0; kt < 8; ++kt) {
      v16bf am = load_a_frag(ma + kt * 32, H_N, lane);
      v16bf ar = load_a_frag(ra + kt * 32, H_N, lane);
#pragma unroll
      for (int j = 0; j < 4; ++j) {
        accZ[j] = wmma_bf16(am, load_b_lds(lz, kt, j, lane), accZ[j]);
        accH[j] = wmma_bf16(ar, load_b_lds(lh, kt, j, lane), accH[j]);
      }
    }
  }

  int nl = lane & 15, half = lane >> 4;
#pragma unroll
  for (int j = 0; j < 4; ++j) {
    int n = cg * 64 + j * 16 + nl;
    float vbz = bz[n], vbh = bh[n];
#pragma unroll
    for (int v = 0; v < 8; ++v) {
      int row = row0 + half * 8 + v;
      float z  = 1.f / (1.f + expf(-(accZ[j][v] + vbz)));
      float ht = tanhf(accH[j][v] + vbh);
      float mp = mprev ? (float)mprev[(size_t)row * H_N + n] : 0.f;
      float m  = (1.f - z) * mp + z * ht;
      mout[(size_t)row * H_N + n] = (bf16_t)m;
    }
  }
}

__global__ __launch_bounds__(256) void gru_r_kernel(
    const bf16_t* __restrict__ xdst, const bf16_t* __restrict__ mnew,
    const bf16_t* __restrict__ WruP, const float* __restrict__ br,
    bf16_t* __restrict__ rmout) {
  extern __shared__ char smem[];
  bf16_t* lw = (bf16_t*)smem;                 // 32KB panel
  int lane = threadIdx.x;
  int wv   = threadIdx.y;
  int cg   = blockIdx.x & 3;
  int row0 = ((blockIdx.x >> 2) * 8 + wv) * 16;
  unsigned lwA = (unsigned)(uintptr_t)(void*)lw;

  v8f acc[4];
#pragma unroll
  for (int j = 0; j < 4; ++j) acc[j] = vzero8f();

  const bf16_t* xa = xdst + (size_t)row0 * H_N;
  const bf16_t* ma = mnew + (size_t)row0 * H_N;

  // Phase 0: Wr rows (K 0..255) against x_dst.
  if (wv == 0) tdm_load_2d(lwA, WruP + (size_t)(cg * 4) * 512, 1024, 8, 4096);
  __builtin_amdgcn_s_wait_tensorcnt(0);
  __syncthreads();
  for (int kt = 0; kt < 8; ++kt) {
    v16bf a = load_a_frag(xa + kt * 32, H_N, lane);
#pragma unroll
    for (int j = 0; j < 4; ++j)
      acc[j] = wmma_bf16(a, load_b_lds(lw, kt, j, lane), acc[j]);
  }

  // Phase 1: Ur rows (K 256..511) against m.
  __syncthreads();
  if (wv == 0)
    tdm_load_2d(lwA, WruP + (size_t)(8 * 16 + cg * 4) * 512, 1024, 8, 4096);
  __builtin_amdgcn_s_wait_tensorcnt(0);
  __syncthreads();
  for (int kt = 0; kt < 8; ++kt) {
    v16bf a = load_a_frag(ma + kt * 32, H_N, lane);
#pragma unroll
    for (int j = 0; j < 4; ++j)
      acc[j] = wmma_bf16(a, load_b_lds(lw, kt, j, lane), acc[j]);
  }

  int nl = lane & 15, half = lane >> 4;
#pragma unroll
  for (int j = 0; j < 4; ++j) {
    int n = cg * 64 + j * 16 + nl;
    float vb = br[n];
#pragma unroll
    for (int v = 0; v < 8; ++v) {
      int row = row0 + half * 8 + v;
      float r  = 1.f / (1.f + expf(-(acc[j][v] + vb)));
      float mv = (float)mnew[(size_t)row * H_N + n];
      rmout[(size_t)row * H_N + n] = (bf16_t)(r * mv);
    }
  }
}

// ---------------------------------------------------------------------------
// Output-head hidden GEMMs (one-shot; B panels are L2-resident).
// ---------------------------------------------------------------------------

__global__ __launch_bounds__(256) void hidden_q_kernel(
    const bf16_t* __restrict__ mfwd, const bf16_t* __restrict__ tvb,
    const bf16_t* __restrict__ WwP, const float* __restrict__ bW,
    bf16_t* __restrict__ qh) {
  int lane = threadIdx.x;
  int w    = blockIdx.x * 8 + threadIdx.y;
  int row0 = (w >> 2) * 16;
  int cg   = w & 3;
  int t    = row0 >> 12;
  int b0   = row0 & (B_N - 1);
  v8f acc[4];
#pragma unroll
  for (int j = 0; j < 4; ++j) acc[j] = vzero8f();

  if (t > 0) {
    const bf16_t* ha = mfwd + ((size_t)(t - 1) * B_N + b0) * H_N;
    for (int kt = 0; kt < 8; ++kt) {
      v16bf a = load_a_frag(ha + kt * 32, H_N, lane);
#pragma unroll
      for (int j = 0; j < 4; ++j)
        acc[j] = wmma_bf16(a, load_b_frag(WwP, kt, cg * 4 + j, 16, lane), acc[j]);
    }
  }
  const bf16_t* ta = tvb + (size_t)b0 * LAT_N;
  for (int kt = 8; kt < 10; ++kt) {
    v16bf a = load_a_frag(ta + (kt - 8) * 32, LAT_N, lane);
#pragma unroll
    for (int j = 0; j < 4; ++j)
      acc[j] = wmma_bf16(a, load_b_frag(WwP, kt, cg * 4 + j, 16, lane), acc[j]);
  }
  int nl = lane & 15, half = lane >> 4;
#pragma unroll
  for (int j = 0; j < 4; ++j) {
    int n = cg * 64 + j * 16 + nl;
    float vb = bW[n];
#pragma unroll
    for (int v = 0; v < 8; ++v) {
      int row = row0 + half * 8 + v;
      float val = acc[j][v] + vb;
      qh[(size_t)row * H_N + n] = (bf16_t)fmaxf(val, 0.f);
    }
  }
}

__global__ __launch_bounds__(256) void hidden_p_kernel(
    const bf16_t* __restrict__ x, const bf16_t* __restrict__ mfwd,
    const bf16_t* __restrict__ hrev, const bf16_t* __restrict__ tvb,
    const bf16_t* __restrict__ UwP, const float* __restrict__ bU,
    bf16_t* __restrict__ ph) {
  int lane = threadIdx.x;
  int w    = blockIdx.x * 8 + threadIdx.y;
  int row0 = (w >> 2) * 16;
  int cg   = w & 3;
  int s    = row0 >> 12;
  int b0   = row0 & (B_N - 1);
  v8f acc[4];
#pragma unroll
  for (int j = 0; j < 4; ++j) acc[j] = vzero8f();

  int tx = (s == 0) ? 0 : ((s <= 15) ? s : 30 - s);
  const bf16_t* xa = x + ((size_t)tx * B_N + b0) * H_N;
  for (int kt = 0; kt < 8; ++kt) {
    v16bf a = load_a_frag(xa + kt * 32, H_N, lane);
#pragma unroll
    for (int j = 0; j < 4; ++j)
      acc[j] = wmma_bf16(a, load_b_frag(UwP, kt, cg * 4 + j, 16, lane), acc[j]);
  }
  const bf16_t* hsrc = nullptr;
  if (s >= 1 && s <= 15)      hsrc = mfwd + ((size_t)(s - 1) * B_N + b0) * H_N;
  else if (s >= 16)           hsrc = hrev + ((size_t)(30 - s) * B_N + b0) * H_N;
  if (hsrc) {
    for (int kt = 8; kt < 16; ++kt) {
      v16bf a = load_a_frag(hsrc + (kt - 8) * 32, H_N, lane);
#pragma unroll
      for (int j = 0; j < 4; ++j)
        acc[j] = wmma_bf16(a, load_b_frag(UwP, kt, cg * 4 + j, 16, lane), acc[j]);
    }
  }
  const bf16_t* ta = tvb + (size_t)b0 * LAT_N;
  for (int kt = 16; kt < 18; ++kt) {
    v16bf a = load_a_frag(ta + (kt - 16) * 32, LAT_N, lane);
#pragma unroll
    for (int j = 0; j < 4; ++j)
      acc[j] = wmma_bf16(a, load_b_frag(UwP, kt, cg * 4 + j, 16, lane), acc[j]);
  }
  int nl = lane & 15, half = lane >> 4;
#pragma unroll
  for (int j = 0; j < 4; ++j) {
    int n = cg * 64 + j * 16 + nl;
    float vb = bU[n];
#pragma unroll
    for (int v = 0; v < 8; ++v) {
      int row = row0 + half * 8 + v;
      float val = acc[j][v] + vb;
      ph[(size_t)row * H_N + n] = (bf16_t)fmaxf(val, 0.f);
    }
  }
}

// ---------------------------------------------------------------------------
// q logits + logsumexp/argmax loss.
// ---------------------------------------------------------------------------
__global__ __launch_bounds__(256) void q_logits_kernel(
    const bf16_t* __restrict__ qh, const bf16_t* __restrict__ WoP,
    const float* __restrict__ bo, const int* __restrict__ wid,
    float* __restrict__ out) {
  __shared__ float lds[16][808];
  int lane = threadIdx.x;
  int wv   = threadIdx.y;
  int row0 = blockIdx.x * 16;

  v16bf afr[8];
#pragma unroll
  for (int kt = 0; kt < 8; ++kt)
    afr[kt] = load_a_frag(qh + (size_t)row0 * H_N + kt * 32, H_N, lane);

  int nl = lane & 15, half = lane >> 4;
  for (int nt = wv; nt < 50; nt += 8) {
    v8f acc = vzero8f();
#pragma unroll
    for (int kt = 0; kt < 8; ++kt)
      acc = wmma_bf16(afr[kt], load_b_frag(WoP, kt, nt, 50, lane), acc);
    int n = nt * 16 + nl;
    float bias = bo[n];
#pragma unroll
    for (int v = 0; v < 8; ++v) lds[half * 8 + v][n] = acc[v] + bias;
  }
  __syncthreads();

  int tid = wv * 32 + lane;
  int r   = tid >> 4;
  int l16 = tid & 15;
  int row = row0 + r;
  int t   = row >> 12;
  int b   = row & (B_N - 1);
  int qt  = wid[b * L_N + t];

  float mx = -3.0e38f; int amx = 0;
  for (int c = l16; c < V_N; c += 16) {
    float v = lds[r][c];
    if (v > mx) { mx = v; amx = c; }
  }
  for (int d = 8; d >= 1; d >>= 1) {
    float om = __shfl_xor(mx, d, 32);
    int   oa = __shfl_xor(amx, d, 32);
    if (om > mx || (om == mx && oa < amx)) { mx = om; amx = oa; }
  }
  float sum = 0.f;
  for (int c = l16; c < V_N; c += 16) sum += expf(lds[r][c] - mx);
  for (int d = 8; d >= 1; d >>= 1) sum += __shfl_xor(sum, d, 32);

  if (l16 == 0) {
    float logZ = mx + logf(sum);
    float qsel = lds[r][qt];
    atomicAdd(&out[0], logZ - qsel);
    atomicAdd(&out[2], (amx == qt) ? 1.f : 0.f);
  }
}

__global__ __launch_bounds__(256) void p_head_kernel(
    const bf16_t* __restrict__ ph, const float* __restrict__ Us,
    const float* __restrict__ Usb, float* __restrict__ out) {
  int lane = threadIdx.x;
  int row  = blockIdx.x * 8 + threadIdx.y;
  const bf16_t* p = ph + (size_t)row * H_N + lane * 8;
  v8bf vv = *(const v8bf*)p;
  float d = 0.f;
#pragma unroll
  for (int i = 0; i < 8; ++i) d += (float)vv[i] * Us[lane * 8 + i];
  for (int m = 16; m >= 1; m >>= 1) d += __shfl_xor(d, m, 32);
  if (lane == 0) {
    float pv = d + Usb[0];
    int   s  = row >> 12;
    float pt = (s < L_N - 1) ? 1.f : 0.f;
    float loss = fmaxf(pv, 0.f) + log1pf(expf(-fabsf(pv))) - pv * pt;
    int pred = (pv > 0.f) ? 1 : 0;
    atomicAdd(&out[1], loss);
    atomicAdd(&out[3], (pred == (int)pt) ? 1.f : 0.f);
  }
}

// ---------------------------------------------------------------------------
// Host orchestration
// ---------------------------------------------------------------------------
extern "C" void kernel_launch(void* const* d_in, const int* in_sizes, int n_in,
                              void* d_out, int out_size, void* d_ws, size_t ws_size,
                              hipStream_t stream) {
  (void)in_sizes; (void)n_in; (void)out_size; (void)ws_size;
  const int*   wid  = (const int*)  d_in[0];
  const float* tv   = (const float*)d_in[1];
  const float* emb  = (const float*)d_in[2];
  const float* Wz_w = (const float*)d_in[3];
  const float* Wz_b = (const float*)d_in[4];
  const float* Wr_w = (const float*)d_in[5];
  const float* Ur_w = (const float*)d_in[6];
  const float* Ur_b = (const float*)d_in[7];
  const float* Wh_w = (const float*)d_in[8];
  const float* Wh_b = (const float*)d_in[9];
  const float* W_w  = (const float*)d_in[10];
  const float* W_b  = (const float*)d_in[11];
  const float* U_w  = (const float*)d_in[12];
  const float* U_b  = (const float*)d_in[13];
  const float* Wo_w = (const float*)d_in[14];
  const float* Wo_b = (const float*)d_in[15];
  const float* Us_w = (const float*)d_in[16];
  const float* Us_b = (const float*)d_in[17];
  float* out = (float*)d_out;

  char* ws = (char*)d_ws;
  size_t off = 0;
  auto alloc = [&](size_t bytes) -> char* {
    off = (off + 255) & ~(size_t)255;
    char* p = ws + off; off += bytes; return p;
  };
  bf16_t* x_bf  = (bf16_t*)alloc((size_t)L_N * STEP * 2);
  bf16_t* tv_bf = (bf16_t*)alloc((size_t)B_N * LAT_N * 2);
  bf16_t* WzP   = (bf16_t*)alloc((size_t)512 * 256 * 2);
  bf16_t* WhP   = (bf16_t*)alloc((size_t)512 * 256 * 2);
  bf16_t* WruP  = (bf16_t*)alloc((size_t)512 * 256 * 2);
  bf16_t* WwP   = (bf16_t*)alloc((size_t)320 * 256 * 2);
  bf16_t* UwP   = (bf16_t*)alloc((size_t)576 * 256 * 2);
  bf16_t* WoP   = (bf16_t*)alloc((size_t)256 * 800 * 2);
  bf16_t* m_fwd = (bf16_t*)alloc((size_t)15 * STEP * 2);
  bf16_t* m_rev = (bf16_t*)alloc((size_t)15 * STEP * 2);
  bf16_t* h_rev = (bf16_t*)alloc((size_t)15 * STEP * 2);
  bf16_t* rm_f  = (bf16_t*)alloc((size_t)STEP * 2);
  bf16_t* rm_r  = (bf16_t*)alloc((size_t)STEP * 2);
  bf16_t* q_hid = (bf16_t*)alloc((size_t)L_N * B_N * H_N * 2);
  bf16_t* p_hid = (bf16_t*)alloc((size_t)(2 * L_N - 1) * B_N * H_N * 2);

  init_out_kernel<<<1, 32, 0, stream>>>(out);

  auto pack = [&](const float* src, bf16_t* dst, int N, int K) {
    int total = N * K;
    pack_w_kernel<<<(total + 255) / 256, 256, 0, stream>>>(src, dst, N, K);
  };
  pack(Wz_w, WzP, 256, 512);
  pack(Wh_w, WhP, 256, 512);
  pack(Wr_w, WruP, 256, 256);
  pack(Ur_w, WruP + (size_t)256 * 256, 256, 256);
  pack(W_w,  WwP, 256, 320);
  pack(U_w,  UwP, 256, 576);
  pack(Wo_w, WoP, 800, 256);

  embed_kernel<<<(L_N * B_N * H_N) / 256, 256, 0, stream>>>(wid, emb, x_bf);
  cvt_kernel<<<(B_N * LAT_N + 255) / 256, 256, 0, stream>>>(tv, tv_bf, B_N * LAT_N);

  dim3 gb(32, 8);
  const int gruBlocks = 128;

  for (int i = 0; i < 15; ++i) {
    const bf16_t* mp = (i > 0) ? (m_fwd + (size_t)(i - 1) * STEP) : nullptr;
    const bf16_t* rp = (i > 0) ? rm_f : nullptr;
    gru_zh_kernel<<<gruBlocks, gb, 65536, stream>>>(
        x_bf + (size_t)i * STEP, mp, rp, WzP, WhP, Wz_b, Wh_b,
        m_fwd + (size_t)i * STEP);
    if (i < 14)
      gru_r_kernel<<<gruBlocks, gb, 32768, stream>>>(
          x_bf + (size_t)(i + 1) * STEP, m_fwd + (size_t)i * STEP,
          WruP, Ur_b, rm_f);
  }
  for (int i = 0; i < 15; ++i) {
    const bf16_t* mp = (i > 0) ? (m_rev + (size_t)(15 - i) * STEP) : nullptr;
    const bf16_t* rp = (i > 0) ? rm_r : nullptr;
    gru_zh_kernel<<<gruBlocks, gb, 65536, stream>>>(
        x_bf + (size_t)(15 - i) * STEP, mp, rp, WzP, WhP, Wz_b, Wh_b,
        m_rev + (size_t)(14 - i) * STEP);
    if (i < 14)
      gru_r_kernel<<<gruBlocks, gb, 32768, stream>>>(
          x_bf + (size_t)(14 - i) * STEP, m_rev + (size_t)(14 - i) * STEP,
          WruP, Ur_b, rm_r);
  }

  hrev_kernel<<<(15 * STEP) / 256, 256, 0, stream>>>(m_fwd, m_rev, h_rev, 15 * STEP);

  hidden_q_kernel<<<2048, gb, 0, stream>>>(m_fwd, tv_bf, WwP, W_b, q_hid);
  hidden_p_kernel<<<3968, gb, 0, stream>>>(x_bf, m_fwd, h_rev, tv_bf, UwP, U_b, p_hid);

  q_logits_kernel<<<(L_N * B_N) / 16, gb, 0, stream>>>(q_hid, WoP, Wo_b, wid, out);
  p_head_kernel<<<((2 * L_N - 1) * B_N) / 8, gb, 0, stream>>>(p_hid, Us_w, Us_b, out);

  final_kernel<<<1, 32, 0, stream>>>(out);
}
